// DynamicCrossAttention_74079595922083
// MI455X (gfx1250) — matrix operand order repaired and checked by reading, not compiled, not run on hardware
//
#include <hip/hip_runtime.h>
#include <hip/hip_bf16.h>
#include <math.h>

#define DIM 512
#define NB  4
#define NN  4096
#define MM  4096

typedef __bf16 bf16;
typedef bf16  v16bf __attribute__((ext_vector_type(16)));
typedef bf16  v8bf  __attribute__((ext_vector_type(8)));
typedef float v8f   __attribute__((ext_vector_type(8)));

union BF16x16 { v16bf v; v8bf h[2]; };

// ---------------------------------------------------------------------------
// LayerNorm: one wave per 512-elem row, bf16 output for WMMA consumers.
// ---------------------------------------------------------------------------
__global__ void ln_rows(const float* __restrict__ x, const float* __restrict__ g,
                        const float* __restrict__ b, bf16* __restrict__ out, int rows) {
  int wave = threadIdx.x >> 5;
  int lane = threadIdx.x & 31;
  int row  = blockIdx.x * 8 + wave;
  if (row >= rows) return;
  const float* xr = x + (size_t)row * DIM;
  float v[16]; float s = 0.f;
#pragma unroll
  for (int i = 0; i < 16; ++i) { v[i] = xr[lane + 32 * i]; s += v[i]; }
#pragma unroll
  for (int o = 16; o > 0; o >>= 1) s += __shfl_xor(s, o, 32);
  float mu = s * (1.f / DIM);
  float var = 0.f;
#pragma unroll
  for (int i = 0; i < 16; ++i) { float d = v[i] - mu; var += d * d; }
#pragma unroll
  for (int o = 16; o > 0; o >>= 1) var += __shfl_xor(var, o, 32);
  float rstd = rsqrtf(var * (1.f / DIM) + 1e-5f);
  bf16* orow = out + (size_t)row * DIM;
#pragma unroll
  for (int i = 0; i < 16; ++i) {
    int c = lane + 32 * i;
    orow[c] = (bf16)((v[i] - mu) * rstd * g[c] + b[c]);
  }
}

// ---------------------------------------------------------------------------
// Transpose + convert the four 512x512 weight matrices to bf16 (WT[n][k]).
// ---------------------------------------------------------------------------
__global__ void prep_wT(const float* __restrict__ Wq, const float* __restrict__ Wk,
                        const float* __restrict__ Wv, const float* __restrict__ Wp,
                        bf16* __restrict__ out) {
  int idx = blockIdx.x * 256 + threadIdx.x;
  const float* W = (blockIdx.y == 0) ? Wq : (blockIdx.y == 1) ? Wk
                 : (blockIdx.y == 2) ? Wv : Wp;
  bf16* o = out + (size_t)blockIdx.y * DIM * DIM;
  int k = idx >> 9, n = idx & 511;
  o[(size_t)n * DIM + k] = (bf16)W[(size_t)k * DIM + n];
}

// ---------------------------------------------------------------------------
// Generic rows x 512 x 512 bf16 WMMA GEMM:  C = A @ W + bias (+resid)
// Block: 256 thr (8 waves). grid = (rows/16, 4). A tile staged in LDS
// with padded stride 520 (260 dwords) -> conflict-free ds_load_b128.
// ---------------------------------------------------------------------------
__global__ void gemm512_k(const bf16* __restrict__ A, const bf16* __restrict__ WT,
                          const float* __restrict__ bias, float* __restrict__ Cf,
                          bf16* __restrict__ Cbf, const float* __restrict__ resid) {
  __shared__ __align__(16) bf16 As[16 * 520];
  int r0 = blockIdx.x * 16;
  int tid = threadIdx.x;
  const bf16* Arow = A + (size_t)r0 * DIM;
  for (int i = tid; i < 16 * 64; i += 256) {
    int rr = i >> 6, oo = (i & 63) << 3;
    *(v8bf*)(As + rr * 520 + oo) = *(const v8bf*)(Arow + (size_t)rr * DIM + oo);
  }
  __syncthreads();
  int wave = tid >> 5, lane = tid & 31, hh = lane >> 4;
  int n = blockIdx.y * 128 + wave * 16 + (lane & 15);
  const bf16* arow = As + (lane & 15) * 520;
  const bf16* brow = WT + (size_t)n * DIM;
  v8f acc = {};
#pragma unroll
  for (int kb = 0; kb < DIM; kb += 32) {
    BF16x16 a, bm;
    // A operand (16x32 bf16): half-wave hh holds K = {8hh..8hh+7, 16+8hh..23+8hh}
    a.h[0] = *(const v8bf*)(arow + kb + 8 * hh);
    a.h[1] = *(const v8bf*)(arow + kb + 16 + 8 * hh);
    // B operand (32x16 bf16): half-wave hh holds K = {8hh..8hh+7, 16+8hh..23+8hh}
    bm.h[0] = *(const v8bf*)(brow + kb + 8 * hh);
    bm.h[1] = *(const v8bf*)(brow + kb + 16 + 8 * hh);
    acc = __builtin_amdgcn_wmma_f32_16x16x32_bf16(false, a.v, false, bm.v,
                                                  (short)0, acc, false, false);
  }
  float bn = bias[n];
#pragma unroll
  for (int r = 0; r < 8; ++r) {
    int row = r0 + 8 * hh + r;          // D layout: M = r + 8*(lane/16)
    size_t o = (size_t)row * DIM + n;
    float v = acc[r] + bn;
    if (resid) v += resid[o];
    if (Cf)  Cf[o]  = v;
    if (Cbf) Cbf[o] = (bf16)v;
  }
}

// ---------------------------------------------------------------------------
// Threshold MLP: thr = gelu(xn @ Wt1 + bt1) @ Wt2 + bt2   (exact GELU, fp32)
// ---------------------------------------------------------------------------
__global__ void threshold_k(const bf16* __restrict__ xn, const float* __restrict__ Wt1,
                            const float* __restrict__ bt1, const float* __restrict__ Wt2,
                            const float* __restrict__ bt2, float* __restrict__ thr) {
  __shared__ float xs[DIM];
  __shared__ float hs[128];
  int row = blockIdx.x;
  int tid = threadIdx.x;                 // 128 threads
  const bf16* xr = xn + (size_t)row * DIM;
  for (int i = tid; i < DIM; i += 128) xs[i] = (float)xr[i];
  __syncthreads();
  float a = bt1[tid];
  for (int c = 0; c < DIM; ++c) a += xs[c] * Wt1[c * 128 + tid];
  float ge = 0.5f * a * (1.f + erff(a * 0.70710678118654752f));
  hs[tid] = ge * Wt2[tid];
  __syncthreads();
  for (int s = 64; s > 0; s >>= 1) { if (tid < s) hs[tid] += hs[tid + s]; __syncthreads(); }
  if (tid == 0) thr[row] = hs[0] + bt2[0];
}

// ---------------------------------------------------------------------------
// Core: S^T = K_tile x Q^T via WMMA bf16, streaming per-lane top-5 of
// masked scores. Block = 16 query rows, 8 waves each sweep disjoint K rows.
// Lane owns one query row (N = lane%16) -> top-5 select is pure per-lane VALU.
// ---------------------------------------------------------------------------
__global__ void attn_topk_k(const bf16* __restrict__ Qbf, const bf16* __restrict__ Kbf,
                            const float* __restrict__ thr, float* __restrict__ topv,
                            int* __restrict__ topi) {
  __shared__ __align__(16) bf16 Qs[16 * 520];
  __shared__ float cv[8][2][16][5];
  __shared__ int   ci[8][2][16][5];
  int b = blockIdx.y;
  int q0 = blockIdx.x * 16;
  int tid = threadIdx.x;
  const bf16* Qrow = Qbf + ((size_t)b * NN + q0) * DIM;
  for (int i = tid; i < 16 * 64; i += 256) {
    int rr = i >> 6, oo = (i & 63) << 3;
    *(v8bf*)(Qs + rr * 520 + oo) = *(const v8bf*)(Qrow + (size_t)rr * DIM + oo);
  }
  __syncthreads();
  int wave = tid >> 5, lane = tid & 31, hh = lane >> 4, j = lane & 15;
  float t = thr[b * NN + q0 + j];
  const bf16* Kb = Kbf + (size_t)b * MM * DIM;
  const bf16* qrow = Qs + j * 520;
  float tv[5]; int ti[5];
#pragma unroll
  for (int k = 0; k < 5; ++k) { tv[k] = -3.0e38f; ti[k] = 0; }
  const float scale = 0.04419417382415922f;  // 1/sqrt(512)
  for (int it = 0; it < 32; ++it) {
    int m0 = it * 128 + wave * 16;
    const bf16* krow = Kb + (size_t)(m0 + j) * DIM;   // A row: M = lane%16
    if (it + 1 < 32)
      __builtin_prefetch(Kb + (size_t)(m0 + 128 + j) * DIM, 0, 0);
    v8f acc = {};
#pragma unroll
    for (int kb = 0; kb < DIM; kb += 32) {
      BF16x16 a, bm;
      a.h[0] = *(const v8bf*)(krow + kb + 8 * hh);
      a.h[1] = *(const v8bf*)(krow + kb + 16 + 8 * hh);
      bm.h[0] = *(const v8bf*)(qrow + kb + 8 * hh);
      bm.h[1] = *(const v8bf*)(qrow + kb + 16 + 8 * hh);
      acc = __builtin_amdgcn_wmma_f32_16x16x32_bf16(false, a.v, false, bm.v,
                                                    (short)0, acc, false, false);
    }
#pragma unroll
    for (int r = 0; r < 8; ++r) {
      float s = acc[r] * scale;
      s = (s > t) ? s : 0.f;                // mask: attn * (attn > thr)
      int m = m0 + r + 8 * hh;              // D layout: M = r + 8*(lane/16)
      if (s > tv[4]) {
        tv[4] = s; ti[4] = m;
#pragma unroll
        for (int k = 4; k > 0; --k)
          if (tv[k] > tv[k - 1]) {
            float tf = tv[k]; tv[k] = tv[k - 1]; tv[k - 1] = tf;
            int q = ti[k]; ti[k] = ti[k - 1]; ti[k - 1] = q;
          }
      }
    }
  }
#pragma unroll
  for (int k = 0; k < 5; ++k) { cv[wave][hh][j][k] = tv[k]; ci[wave][hh][j][k] = ti[k]; }
  __syncthreads();
  if (tid < 16) {  // merge 8 waves x 2 half-waves x 5 = 80 candidates / row
    float fv[5]; int fi[5];
#pragma unroll
    for (int k = 0; k < 5; ++k) { fv[k] = -3.0e38f; fi[k] = 0; }
    for (int w = 0; w < 8; ++w)
      for (int h2 = 0; h2 < 2; ++h2)
#pragma unroll
        for (int k = 0; k < 5; ++k) {
          float s = cv[w][h2][tid][k]; int m = ci[w][h2][tid][k];
          if (s > fv[4]) {
            fv[4] = s; fi[4] = m;
#pragma unroll
            for (int kk = 4; kk > 0; --kk)
              if (fv[kk] > fv[kk - 1]) {
                float tf = fv[kk]; fv[kk] = fv[kk - 1]; fv[kk - 1] = tf;
                int q = fi[kk]; fi[kk] = fi[kk - 1]; fi[kk - 1] = q;
              }
          }
        }
    size_t base = ((size_t)b * NN + q0 + tid) * 5;
#pragma unroll
    for (int k = 0; k < 5; ++k) { topv[base + k] = fv[k]; topi[base + k] = fi[k]; }
  }
}

// ---------------------------------------------------------------------------
// sumV[b][c] = sum_m V[b][m][c]
// ---------------------------------------------------------------------------
__global__ void sumv_k(const float* __restrict__ V, float* __restrict__ sumV) {
  int b = blockIdx.y;
  int c = blockIdx.x * 256 + threadIdx.x;
  const float* vb = V + (size_t)b * MM * DIM + c;
  float s = 0.f;
  for (int m = 0; m < MM; ++m) s += vb[(size_t)m * DIM];
  sumV[b * DIM + c] = s;
}

// ---------------------------------------------------------------------------
// Sparse-softmax recombination: row = (sumV + sum_t (e^v-1) V[i_t]) / Z
// Z = M + sum_t (e^v - 1). Output bf16 (input to final WMMA GEMM).
// ---------------------------------------------------------------------------
__global__ void attnout_k(const float* __restrict__ V, const float* __restrict__ sumV,
                          const float* __restrict__ topv, const int* __restrict__ topi,
                          bf16* __restrict__ ao) {
  int row = blockIdx.x;             // b*N + n
  int b = row >> 12;
  int tid = threadIdx.x;            // 128 threads
  float w[5]; int ix[5]; float Z = (float)MM;
#pragma unroll
  for (int t = 0; t < 5; ++t) {
    float v = topv[(size_t)row * 5 + t];
    ix[t] = topi[(size_t)row * 5 + t];
    w[t] = expf(v) - 1.f;
    Z += w[t];
  }
  float invZ = 1.f / Z;
  const float* Vb = V + (size_t)b * MM * DIM;
  bf16* orow = ao + (size_t)row * DIM;
#pragma unroll
  for (int i = 0; i < 4; ++i) {
    int c = tid + 128 * i;
    float acc = sumV[b * DIM + c];
#pragma unroll
    for (int t = 0; t < 5; ++t) acc += w[t] * Vb[(size_t)ix[t] * DIM + c];
    orow[c] = (bf16)(acc * invZ);
  }
}

// ---------------------------------------------------------------------------
extern "C" void kernel_launch(void* const* d_in, const int* in_sizes, int n_in,
                              void* d_out, int out_size, void* d_ws, size_t ws_size,
                              hipStream_t stream) {
  (void)in_sizes; (void)n_in; (void)out_size; (void)ws_size;
  const float* x   = (const float*)d_in[0];
  const float* ctx = (const float*)d_in[1];
  const float* Wq  = (const float*)d_in[2];
  const float* bq  = (const float*)d_in[3];
  const float* Wk  = (const float*)d_in[4];
  const float* bk  = (const float*)d_in[5];
  const float* Wv  = (const float*)d_in[6];
  const float* bv  = (const float*)d_in[7];
  const float* Wt1 = (const float*)d_in[8];
  const float* bt1 = (const float*)d_in[9];
  const float* Wt2 = (const float*)d_in[10];
  const float* bt2 = (const float*)d_in[11];
  const float* Wp  = (const float*)d_in[12];
  const float* bp  = (const float*)d_in[13];
  const float* g1  = (const float*)d_in[14];
  const float* b1  = (const float*)d_in[15];
  const float* g2  = (const float*)d_in[16];
  const float* b2  = (const float*)d_in[17];
  float* out = (float*)d_out;

  char* p = (char*)d_ws;
  auto carve = [&](size_t bytes) { char* r = p; p += (bytes + 255) & ~(size_t)255; return r; };
  const size_t BN = (size_t)NB * NN, BM = (size_t)NB * MM;
  bf16*  xn  = (bf16*)carve(BN * DIM * 2);
  bf16*  cn  = (bf16*)carve(BM * DIM * 2);
  bf16*  Qb  = (bf16*)carve(BN * DIM * 2);
  bf16*  Kb  = (bf16*)carve(BM * DIM * 2);
  float* Vf  = (float*)carve(BM * DIM * 4);
  bf16*  AO  = (bf16*)carve(BN * DIM * 2);
  bf16*  WT  = (bf16*)carve((size_t)4 * DIM * DIM * 2);
  float* thr = (float*)carve(BN * 4);
  float* tpv = (float*)carve(BN * 5 * 4);
  int*   tpi = (int*)carve(BN * 5 * 4);
  float* sV  = (float*)carve((size_t)NB * DIM * 4);

  bf16* WqT = WT;
  bf16* WkT = WT + (size_t)DIM * DIM;
  bf16* WvT = WT + (size_t)2 * DIM * DIM;
  bf16* WpT = WT + (size_t)3 * DIM * DIM;

  hipLaunchKernelGGL(prep_wT, dim3(DIM * DIM / 256, 4), dim3(256), 0, stream,
                     Wq, Wk, Wv, Wp, WT);
  hipLaunchKernelGGL(ln_rows, dim3((unsigned)(BN / 8)), dim3(256), 0, stream,
                     x, g1, b1, xn, (int)BN);
  hipLaunchKernelGGL(ln_rows, dim3((unsigned)(BM / 8)), dim3(256), 0, stream,
                     ctx, g2, b2, cn, (int)BM);
  hipLaunchKernelGGL(gemm512_k, dim3((unsigned)(BN / 16), 4), dim3(256), 0, stream,
                     xn, WqT, bq, (float*)nullptr, Qb, (const float*)nullptr);
  hipLaunchKernelGGL(gemm512_k, dim3((unsigned)(BM / 16), 4), dim3(256), 0, stream,
                     cn, WkT, bk, (float*)nullptr, Kb, (const float*)nullptr);
  hipLaunchKernelGGL(gemm512_k, dim3((unsigned)(BM / 16), 4), dim3(256), 0, stream,
                     cn, WvT, bv, Vf, (bf16*)nullptr, (const float*)nullptr);
  hipLaunchKernelGGL(threshold_k, dim3((unsigned)BN), dim3(128), 0, stream,
                     xn, Wt1, bt1, Wt2, bt2, thr);
  hipLaunchKernelGGL(attn_topk_k, dim3(NN / 16, NB), dim3(256), 0, stream,
                     Qb, Kb, thr, tpv, tpi);
  hipLaunchKernelGGL(sumv_k, dim3(2, NB), dim3(256), 0, stream, Vf, sV);
  hipLaunchKernelGGL(attnout_k, dim3((unsigned)BN), dim3(128), 0, stream,
                     Vf, sV, tpv, tpi, AO);
  hipLaunchKernelGGL(gemm512_k, dim3((unsigned)(BN / 16), 4), dim3(256), 0, stream,
                     AO, WpT, bp, out, (bf16*)nullptr, x);
}
